// DampedOscillator_50148038148154
// MI455X (gfx1250) — matrix operand order repaired and checked by reading, not compiled, not run on hardware
//
#include <hip/hip_runtime.h>
#include <hip/hip_bf16.h>

typedef __attribute__((ext_vector_type(16))) _Float16 v16h;
typedef __attribute__((ext_vector_type(8)))  _Float16 v8h;
typedef __attribute__((ext_vector_type(8)))  float    v8f;

#define A_NUM    16
#define M_NUM    64
#define T_NUM    32768
#define F_LEN    4096
#define BINS     64
#define FSH_LEN  4608            /* per (audio,shift) force row: kb+p max 4351 < 4608 */
#define SPAD_L   4095            /* left zero pad of signal (odd on purpose: aligns A) */
#define SPAD_LEN 37120           /* 4095 + 32768 + pad; multiple of 32 halfs */
#define TWO_PI   6.28318530717958647692f
#define LN10     2.30258509299404568402f

// ---------------------------------------------------------------------------
// Kernel 1: per-mode damp/freq + per-(audio,mode) amp (f16). One block, 1024 thr.
// ---------------------------------------------------------------------------
__global__ void prep_kernel(const float* __restrict__ freq_linear,
                            const float* __restrict__ amp_value,
                            const float* __restrict__ alpha_params,
                            const float* __restrict__ beta_params,
                            float* __restrict__ damp,
                            float* __restrict__ freqd,
                            _Float16* __restrict__ amp_h)
{
    int tid = threadIdx.x;
    if (tid < M_NUM) {
        int m = tid;
        float an = 0.f, ad = 0.f, bn = 0.f, bd = 0.f;
        const float la0 = logf(6.0f);
        const float lb0 = logf(2e-7f);
        for (int b = 0; b < BINS; ++b) {
            float frac = -1.0f + 2.0f * (float)b / 63.0f;   // log10-spaced c/10..c*10
            float av = __expf(la0 + frac * LN10);
            float bv = __expf(lb0 + frac * LN10);
            float sa = log1pf(__expf(alpha_params[m * BINS + b]));  // softplus
            float sb = log1pf(__expf(beta_params[m * BINS + b]));
            an += sa * av;  ad += sa;
            bn += sb * bv;  bd += sb;
        }
        float alpha = an / ad;
        float beta  = bn / bd;
        float w   = TWO_PI * freq_linear[m];
        float lbd = w * w;
        float dv  = 0.5f * (alpha + beta * lbd);
        damp[m]  = dv;
        freqd[m] = sqrtf(lbd - dv * dv) * (1.0f / TWO_PI);
    }
    {   // amp for all 16*64 entries (modified sigmoid)
        float x = amp_value[tid];
        float s = 1.0f / (1.0f + __expf(-x));
        float amp = 2.0f * powf(s, LN10) + 1e-7f;
        amp_h[tid] = (_Float16)amp;
    }
}

// ---------------------------------------------------------------------------
// Kernel 2: 16 pre-shifted, zero-padded f16 copies of each force vector:
//   fsh[a][c][u] = forces[a][u - c]  (zero outside [0,4096)),  u in [0,4608)
// Gives 32B-aligned contiguous B-operand windows in the conv loop.
// ---------------------------------------------------------------------------
__global__ void fsh_kernel(const float* __restrict__ forces,
                           _Float16* __restrict__ fsh)
{
    int idx = blockIdx.x * blockDim.x + threadIdx.x;  // a*16*4608 + c*4608 + u
    if (idx >= A_NUM * 16 * FSH_LEN) return;
    int u  = idx % FSH_LEN;
    int ac = idx / FSH_LEN;
    int c  = ac & 15;
    int a  = ac >> 4;
    int k  = u - c;
    float v = (k >= 0 && k < F_LEN) ? forces[a * F_LEN + k] : 0.0f;
    fsh[idx] = (_Float16)v;
}

// ---------------------------------------------------------------------------
// Kernel 3: zero the padded-signal buffer (u32 = pairs of halfs)
// ---------------------------------------------------------------------------
__global__ void zero_spad_kernel(unsigned int* __restrict__ p, int n)
{
    int i = blockIdx.x * blockDim.x + threadIdx.x;
    if (i < n) p[i] = 0u;
}

// ---------------------------------------------------------------------------
// Kernel 4: E matrix, transposed layout Et[t][m] (f16)
// ---------------------------------------------------------------------------
__global__ void emat_kernel(const float* __restrict__ damp,
                            const float* __restrict__ freqd,
                            _Float16* __restrict__ Et)
{
    int idx = blockIdx.x * blockDim.x + threadIdx.x;   // idx = t*64 + m
    int t = idx >> 6, m = idx & 63;
    float tt = (float)(t + 1) * (1.0f / 16000.0f);
    float e = __expf(-damp[m] * tt) * __sinf(TWO_PI * freqd[m] * tt);
    Et[idx] = (_Float16)e;
}

// ---------------------------------------------------------------------------
// Kernel 5: signal = Amp(16x64) x E(64xT) via v_wmma_f32_16x16x32_f16.
// One wave per 16-column t-tile (2048 waves). Writes f16 signal at pad 4095.
// ---------------------------------------------------------------------------
__global__ void __launch_bounds__(128)
signal_gemm_kernel(const _Float16* __restrict__ amp_h,
                   const _Float16* __restrict__ Et,
                   _Float16* __restrict__ spad)
{
    int lane = threadIdx.x & 31;
    int wid  = blockIdx.x * (blockDim.x >> 5) + (threadIdx.x >> 5);
    int t0 = wid * 16;
    int hi = lane >> 4;
    int q  = lane & 15;

    // A operand: Amp rows=audio(q), K=mode.  16-bit A 16x32 layout.
    v16h a0, a1;
#pragma unroll
    for (int j = 0; j < 8; ++j) {
        int k = hi * 16 + 2 * j;
        a0[2 * j]     = amp_h[q * M_NUM + k];
        a0[2 * j + 1] = amp_h[q * M_NUM + k + 1];
        a1[2 * j]     = amp_h[q * M_NUM + 32 + k];
        a1[2 * j + 1] = amp_h[q * M_NUM + 32 + k + 1];
    }
    // B operand: B[K][c] = Et[t0+c][kc + hi*16 + e] -> aligned v16h loads
    v16h b0 = *(const v16h*)(Et + (t0 + q) * M_NUM + hi * 16);
    v16h b1 = *(const v16h*)(Et + (t0 + q) * M_NUM + 32 + hi * 16);

    v8f c = {};
    c = __builtin_amdgcn_wmma_f32_16x16x32_f16(false, a0, false, b0, (short)0, c, false, false);
    c = __builtin_amdgcn_wmma_f32_16x16x32_f16(false, a1, false, b1, (short)0, c, false, false);

    // C layout: vgpr v -> row M = v + hi*8 (audio), col N = q (t offset)
#pragma unroll
    for (int v = 0; v < 8; ++v) {
        int row = v + hi * 8;
        spad[row * SPAD_LEN + SPAD_L + t0 + q] = (_Float16)c[v];
    }
}

// ---------------------------------------------------------------------------
// Kernel 6: depthwise long conv as WMMA (signal in A, shifted forces in B).
//   C[r][c] += sum_p S[t0+16r+kb+p-4095] * F[kb+p-c]
//   signal_index - tap = t0+16r+c-4095+4095 -> out position t0+16r+c (p-free).
// For fixed c, taps kb+p-c over {kb=0..4320 step 32} x {p=0..31} cover each
// k in [0,4096) exactly once; zero pads handle out-of-range taps/signal.
// Inner loop: 4 aligned b128 loads + 1 WMMA, fully branchless (EXEC all-1s).
// ---------------------------------------------------------------------------
__global__ void __launch_bounds__(128)
conv_wmma_kernel(const _Float16* __restrict__ fsh,
                 const _Float16* __restrict__ spad,
                 float* __restrict__ out)
{
    int lane = threadIdx.x & 31;
    int wid  = blockIdx.x * (blockDim.x >> 5) + (threadIdx.x >> 5);
    int a  = wid >> 7;          // audio 0..15
    int tb = wid & 127;         // 128 t-blocks of 256 samples
    int t0 = tb << 8;
    int hi = lane >> 4;
    int q  = lane & 15;         // A row r / B column c

    // A: spad[a][t0 + 16q + kb + p], p-chunks hi*16..+7 and hi*16+8..+7
    //    index range [0, 37104) always inside SPAD_LEN; 16B-aligned chunks.
    const _Float16* Ap = spad + a * SPAD_LEN + t0 + 16 * q + hi * 16;
    // B: fsh[a][q][kb + hi*16 + e], 16 contiguous halfs, 32B-aligned.
    const _Float16* Bp = fsh + (a * 16 + q) * FSH_LEN + hi * 16;

    v8f c = {};
#pragma unroll 2
    for (int kb = 0; kb <= 4320; kb += 32) {
        v8h alo = *(const v8h*)(Ap + kb);
        v8h ahi = *(const v8h*)(Ap + kb + 8);
        v16h av = __builtin_shufflevector(alo, ahi,
                    0, 1, 2, 3, 4, 5, 6, 7, 8, 9, 10, 11, 12, 13, 14, 15);
        v16h bv = *(const v16h*)(Bp + kb);
        c = __builtin_amdgcn_wmma_f32_16x16x32_f16(false, av, false, bv, (short)0, c, false, false);
    }

    float* op = out + a * T_NUM + t0 + q;
#pragma unroll
    for (int v = 0; v < 8; ++v) {
        int row = v + hi * 8;
        op[16 * row] = c[v];
    }
}

// ---------------------------------------------------------------------------
extern "C" void kernel_launch(void* const* d_in, const int* in_sizes, int n_in,
                              void* d_out, int out_size, void* d_ws, size_t ws_size,
                              hipStream_t stream)
{
    const float* freq_linear  = (const float*)d_in[0];   // (64)
    const float* amp_value    = (const float*)d_in[1];   // (16,64,1)
    const float* alpha_params = (const float*)d_in[2];   // (1,64,1,64)
    const float* beta_params  = (const float*)d_in[3];   // (1,64,1,64)
    const float* forces       = (const float*)d_in[4];   // (16,1,4096)
    float* out = (float*)d_out;                          // (16,32768)

    char* ws = (char*)d_ws;
    float*    damp  = (float*)(ws + 0);             // 64 f32
    float*    freqd = (float*)(ws + 1024);          // 64 f32
    _Float16* amp_h = (_Float16*)(ws + 2048);       // 16*64 f16    (ends 4096)
    _Float16* fsh   = (_Float16*)(ws + 4096);       // 16*16*4608 f16 = 2359296B (ends 2363392)
    _Float16* Et    = (_Float16*)(ws + 2363392);    // 32768*64 f16 = 4194304B  (ends 6557696)
    _Float16* spad  = (_Float16*)(ws + 6557696);    // 16*37120 f16 = 1187840B  (ends 7745536)

    prep_kernel<<<1, 1024, 0, stream>>>(freq_linear, amp_value, alpha_params,
                                        beta_params, damp, freqd, amp_h);

    fsh_kernel<<<(A_NUM * 16 * FSH_LEN + 255) / 256, 256, 0, stream>>>(forces, fsh);

    int spad_u32 = A_NUM * SPAD_LEN / 2;   // 296960
    zero_spad_kernel<<<(spad_u32 + 255) / 256, 256, 0, stream>>>(
        (unsigned int*)spad, spad_u32);

    emat_kernel<<<(T_NUM * M_NUM) / 256, 256, 0, stream>>>(damp, freqd, Et);

    // 2048 tiles, 4 waves/block of 128 threads -> 512 blocks, exact (EXEC all-1s)
    signal_gemm_kernel<<<512, 128, 0, stream>>>(amp_h, Et, spad);

    // 16 audios x 128 t-blocks = 2048 waves -> 512 blocks x 128 threads, exact
    conv_wmma_kernel<<<512, 128, 0, stream>>>(fsh, spad, out);
}
